// NIC_88330297409644
// MI455X (gfx1250) — matrix operand order repaired and checked by reading, not compiled, not run on hardware
//
#include <hip/hip_runtime.h>
#include <hip/hip_bf16.h>
#include <math.h>

// ---------------- CDNA5 WMMA types ----------------
typedef __attribute__((ext_vector_type(2)))  float  v2f;
typedef __attribute__((ext_vector_type(8)))  float  v8f;
typedef __attribute__((ext_vector_type(16))) __bf16 v16bf;

union FragBF16 { uint4 q[2]; v16bf v; };

__device__ __forceinline__ unsigned short f2bf(float x) {
    unsigned int u = __float_as_uint(x);
    unsigned int r = u + 0x7fffu + ((u >> 16) & 1u);   // round-to-nearest-even
    return (unsigned short)(r >> 16);
}
__device__ __forceinline__ float sigm(float x) { return 1.0f / (1.0f + expf(-x)); }

static constexpr int B    = 64;
static constexpr int T    = 20;
static constexpr int DIN  = 512;
static constexpr int DR   = 512;
static constexpr int DF   = 2048;
static constexpr int V    = 32001;
static constexpr int VPAD = 32064;   // pad vocab to multiple of 64

// ---------------- small prep kernels ----------------
__global__ void __launch_bounds__(256) prep_bias(const float* __restrict__ b_ih,
                                                 const float* __restrict__ b_hh,
                                                 float* __restrict__ bias) {
    int i = blockIdx.x * 256 + threadIdx.x;
    if (i < 4 * DR) bias[i] = b_ih[i] + b_hh[i];
}

__global__ void __launch_bounds__(256) conv_wlogit(const float* __restrict__ W,
                                                   ushort* __restrict__ Wb) {
    const int total4 = (VPAD * DR) >> 2;                 // process 4 elems/thread
    for (int i4 = blockIdx.x * blockDim.x + threadIdx.x; i4 < total4;
         i4 += gridDim.x * blockDim.x) {
        int i = i4 << 2;
        int row = i >> 9;                                // / DR
        ushort4 o;
        if (row < V) {
            float4 f = *(const float4*)(W + i);
            o.x = f2bf(f.x); o.y = f2bf(f.y); o.z = f2bf(f.z); o.w = f2bf(f.w);
        } else {
            o.x = o.y = o.z = o.w = 0;
        }
        *(ushort4*)(Wb + i) = o;
    }
}

__global__ void __launch_bounds__(128) embed_gather(const int* __restrict__ seq,
                                                    const float* __restrict__ embed,
                                                    float* __restrict__ xs) {
    int row = blockIdx.x;                 // 0..(19*64-1)
    int t = 1 + row / B;
    int b = row % B;
    int tok = seq[b * T + t];
    const float4* src = (const float4*)(embed + (size_t)tok * DIN);
    float4* dst = (float4*)(xs + (size_t)(t * B + b) * DIN);
    dst[threadIdx.x] = src[threadIdx.x];  // 128 threads * 16B = 2KB row
}

// ================= fp32 WMMA GEMM: C = A(MxK) * W(NxK)^T + bias[n] =================
// 8 waves/block; wave computes 16x64 strip (4 tiles). Software-pipelined: while
// chunk i is consumed by v_wmma, chunk i+1's global_load_b64s are in flight.
struct ChunkF32 { v2f a[4]; v2f b[4][4]; };     // 4 k-steps (K=16) for 4 N-tiles

__device__ __forceinline__ void load_chunk_f32(ChunkF32& ch,
                                               const float* __restrict__ aRow,
                                               const float* __restrict__ wRow,
                                               size_t ldb, int k0) {
#pragma unroll
    for (int kk = 0; kk < 4; ++kk) {
        ch.a[kk] = *(const v2f*)(aRow + k0 + 4 * kk);
#pragma unroll
        for (int tn = 0; tn < 4; ++tn)
            ch.b[kk][tn] = *(const v2f*)(wRow + (size_t)(16 * tn) * ldb + k0 + 4 * kk);
    }
}

__device__ __forceinline__ void mma_chunk_f32(v8f acc[4], const ChunkF32& ch) {
#pragma unroll
    for (int kk = 0; kk < 4; ++kk)
#pragma unroll
        for (int tn = 0; tn < 4; ++tn)
            acc[tn] = __builtin_amdgcn_wmma_f32_16x16x4_f32(
                false, ch.a[kk], false, ch.b[kk][tn], (short)0, acc[tn], false, false);
}

__global__ void __launch_bounds__(256) gemm_f32_wmma(const float* __restrict__ A,
                                                     const float* __restrict__ W,
                                                     const float* __restrict__ bias,
                                                     float* __restrict__ C,
                                                     int M, int N, int K,
                                                     int lda, int ldb, int ldc) {
    const int lane = threadIdx.x & 31;
    const int wave = threadIdx.x >> 5;
    const int half = lane >> 4;
    const int l16  = lane & 15;
    const int nStrips = N >> 6;
    const int strip = blockIdx.x * 8 + wave;
    if (strip >= (M >> 4) * nStrips) return;     // whole wave exits -> EXEC stays full
    const int m0 = (strip / nStrips) << 4;
    const int n0 = (strip % nStrips) << 6;

    v8f acc[4] = {{}, {}, {}, {}};
    const float* aRow = A + (size_t)(m0 + l16) * lda + 2 * half;
    const float* wRow = W + (size_t)(n0 + l16) * ldb + 2 * half;

    ChunkF32 ch0, ch1;
    load_chunk_f32(ch0, aRow, wRow, ldb, 0);
#pragma unroll 1
    for (int k0 = 0; k0 < K; k0 += 32) {         // K is a multiple of 32
        load_chunk_f32(ch1, aRow, wRow, ldb, k0 + 16);
        mma_chunk_f32(acc, ch0);
        if (k0 + 32 < K) load_chunk_f32(ch0, aRow, wRow, ldb, k0 + 32);
        mma_chunk_f32(acc, ch1);
    }

#pragma unroll
    for (int r = 0; r < 8; ++r) {
        size_t row = (size_t)(m0 + r + 8 * half) * ldc;
#pragma unroll
        for (int tn = 0; tn < 4; ++tn)
            C[row + n0 + 16 * tn + l16] = acc[tn][r] + bias[n0 + 16 * tn + l16];
    }
}

// ================= LSTM step: gates = h_prev @ W_hh^T + gates_x[t] =================
// 8 blocks x 256 threads; block owns 64 h-columns (all four gates for those cols),
// wave = (m-tile 0..3) x (jhalf 0..1); 8 accum tiles/wave, pipelined k-chunks.
struct ChunkHH { v2f a[2]; v2f b[2][8]; };      // 2 k-steps (K=8) for 8 tiles

__device__ __forceinline__ void load_chunk_hh(ChunkHH& ch,
                                              const float* __restrict__ aRow,
                                              const float* const wBase[8], int k0) {
#pragma unroll
    for (int kk = 0; kk < 2; ++kk) {
        ch.a[kk] = *(const v2f*)(aRow + k0 + 4 * kk);
#pragma unroll
        for (int u = 0; u < 8; ++u)
            ch.b[kk][u] = *(const v2f*)(wBase[u] + k0 + 4 * kk);
    }
}

__device__ __forceinline__ void mma_chunk_hh(v8f acc[8], const ChunkHH& ch) {
#pragma unroll
    for (int kk = 0; kk < 2; ++kk)
#pragma unroll
        for (int u = 0; u < 8; ++u)
            acc[u] = __builtin_amdgcn_wmma_f32_16x16x4_f32(
                false, ch.a[kk], false, ch.b[kk][u], (short)0, acc[u], false, false);
}

__global__ void __launch_bounds__(256) lstm_step(const float* __restrict__ h_prev,
                                                 float* __restrict__ c,
                                                 const float* __restrict__ gx,
                                                 const float* __restrict__ W_hh,
                                                 float* __restrict__ h_new,
                                                 ushort* __restrict__ hs_bf) {
    const int lane  = threadIdx.x & 31;
    const int wave  = threadIdx.x >> 5;
    const int half  = lane >> 4;
    const int l16   = lane & 15;
    const int mt    = wave & 3;
    const int jhalf = wave >> 2;
    const int m0    = mt << 4;
    const int j0    = blockIdx.x << 6;

    v8f acc[8] = {{}, {}, {}, {}, {}, {}, {}, {}};   // u = g*2 + jj
    const float* aRow = h_prev + (size_t)(m0 + l16) * DR + 2 * half;
    const float* wBase[8];
#pragma unroll
    for (int g = 0; g < 4; ++g)
#pragma unroll
        for (int jj = 0; jj < 2; ++jj) {
            int n = g * DR + j0 + (jhalf * 2 + jj) * 16 + l16;
            wBase[g * 2 + jj] = W_hh + (size_t)n * DR + 2 * half;
        }

    ChunkHH ch0, ch1;
    load_chunk_hh(ch0, aRow, wBase, 0);
#pragma unroll 1
    for (int k0 = 0; k0 < DR; k0 += 16) {
        load_chunk_hh(ch1, aRow, wBase, k0 + 8);
        mma_chunk_hh(acc, ch0);
        if (k0 + 16 < DR) load_chunk_hh(ch0, aRow, wBase, k0 + 16);
        mma_chunk_hh(acc, ch1);
    }

#pragma unroll
    for (int jj = 0; jj < 2; ++jj) {
        int jcol = j0 + (jhalf * 2 + jj) * 16 + l16;
#pragma unroll
        for (int r = 0; r < 8; ++r) {
            int m = m0 + r + 8 * half;
            size_t gb = (size_t)m * (4 * DR);
            float iv = acc[0 * 2 + jj][r] + gx[gb            + jcol];
            float fv = acc[1 * 2 + jj][r] + gx[gb +     DR   + jcol];
            float gv = acc[2 * 2 + jj][r] + gx[gb + 2 * DR   + jcol];
            float ov = acc[3 * 2 + jj][r] + gx[gb + 3 * DR   + jcol];
            size_t hi = (size_t)m * DR + jcol;
            float cn = sigm(fv) * c[hi] + sigm(iv) * tanhf(gv);
            float hn = sigm(ov) * tanhf(cn);
            c[hi] = cn;
            h_new[hi] = hn;
            hs_bf[hi] = f2bf(hn);
        }
    }
}

// ======= logits: out[b,t,v] = hs_bf16 @ Wlogit_bf16^T + b_logit (bf16 WMMA) =======
struct ChunkBF { FragBF16 fa; FragBF16 fb[4]; };     // one K=32 step, 4 N-tiles

__device__ __forceinline__ void load_chunk_bf(ChunkBF& ch,
                                              const ushort* __restrict__ aRow,
                                              const ushort* __restrict__ wRow,
                                              int half, int k0) {
    ch.fa.q[0] = *(const uint4*)(aRow + k0 + half * 8);
    ch.fa.q[1] = *(const uint4*)(aRow + k0 + 16 + half * 8);
#pragma unroll
    for (int tn = 0; tn < 4; ++tn) {
        const ushort* wp = wRow + (size_t)tn * 16 * DR + k0 + half * 8;
        ch.fb[tn].q[0] = *(const uint4*)(wp);
        ch.fb[tn].q[1] = *(const uint4*)(wp + 16);
        __builtin_prefetch(wp + 128, 0, 1);      // global_prefetch_b8, stream W ahead
    }
}

__device__ __forceinline__ void mma_chunk_bf(v8f acc[4], const ChunkBF& ch) {
#pragma unroll
    for (int tn = 0; tn < 4; ++tn)
        acc[tn] = __builtin_amdgcn_wmma_f32_16x16x32_bf16(
            false, ch.fa.v, false, ch.fb[tn].v, (short)0, acc[tn], false, false);
}

__global__ void __launch_bounds__(256) logits_gemm(const ushort* __restrict__ hs,
                                                   const ushort* __restrict__ Wb,
                                                   const float* __restrict__ b_logit,
                                                   float* __restrict__ out) {
    const int lane = threadIdx.x & 31;
    const int wave = threadIdx.x >> 5;
    const int half = lane >> 4;
    const int l16  = lane & 15;
    const int nStrips = VPAD >> 6;                       // 501
    const int strip = blockIdx.x * 8 + wave;
    if (strip >= (T * B / 16) * nStrips) return;
    const int m0 = (strip / nStrips) << 4;
    const int n0 = (strip % nStrips) << 6;

    v8f acc[4] = {{}, {}, {}, {}};
    const ushort* aRow = hs + (size_t)(m0 + l16) * DR;
    const ushort* wRow = Wb + (size_t)(n0 + l16) * DR;

    ChunkBF ch0, ch1;
    load_chunk_bf(ch0, aRow, wRow, half, 0);
#pragma unroll 1
    for (int k0 = 0; k0 < DR; k0 += 64) {        // DR multiple of 64
        load_chunk_bf(ch1, aRow, wRow, half, k0 + 32);
        mma_chunk_bf(acc, ch0);
        if (k0 + 64 < DR) load_chunk_bf(ch0, aRow, wRow, half, k0 + 64);
        mma_chunk_bf(acc, ch1);
    }

#pragma unroll
    for (int tn = 0; tn < 4; ++tn) {
        int v = n0 + tn * 16 + l16;
        if (v < V) {
            float bv = b_logit[v];
#pragma unroll
            for (int r = 0; r < 8; ++r) {
                int m = m0 + r + 8 * half;                 // m = t*64 + b
                size_t o = (size_t)(m % B) * (T * V) + (size_t)(m / B) * V + v;
                out[o] = acc[tn][r] + bv;
            }
        }
    }
}

// ---------------- fused log_softmax over each (b,t) row, in place ----------------
__global__ void __launch_bounds__(256) logsoftmax_rows(float* __restrict__ out) {
    __shared__ float red[256];
    const int tid = threadIdx.x;
    float* p = out + (size_t)blockIdx.x * V;

    float mx = -INFINITY;
    for (int i = tid; i < V; i += 256) mx = fmaxf(mx, p[i]);
    red[tid] = mx; __syncthreads();
    for (int s = 128; s > 0; s >>= 1) {
        if (tid < s) red[tid] = fmaxf(red[tid], red[tid + s]);
        __syncthreads();
    }
    mx = red[0]; __syncthreads();

    float sum = 0.0f;
    for (int i = tid; i < V; i += 256) sum += expf(p[i] - mx);
    red[tid] = sum; __syncthreads();
    for (int s = 128; s > 0; s >>= 1) {
        if (tid < s) red[tid] += red[tid + s];
        __syncthreads();
    }
    float lse = mx + logf(red[0]);

    for (int i = tid; i < V; i += 256) p[i] -= lse;
}

// ---------------- host launcher ----------------
extern "C" void kernel_launch(void* const* d_in, const int* in_sizes, int n_in,
                              void* d_out, int out_size, void* d_ws, size_t ws_size,
                              hipStream_t stream) {
    const float* img_feat = (const float*)d_in[0];
    const int*   seq      = (const int*)  d_in[1];
    const float* W_img    = (const float*)d_in[2];
    const float* b_img    = (const float*)d_in[3];
    const float* embed    = (const float*)d_in[4];
    const float* W_ih     = (const float*)d_in[5];
    const float* b_ih     = (const float*)d_in[6];
    const float* W_hh     = (const float*)d_in[7];
    const float* b_hh     = (const float*)d_in[8];
    const float* W_logit  = (const float*)d_in[9];
    const float* b_logit  = (const float*)d_in[10];
    float* out = (float*)d_out;

    char* ws = (char*)d_ws;
    size_t off = 0;
    float* xs       = (float*)(ws + off); off += (size_t)T * B * DIN * 4;       // 2.6 MB
    float* gates_x  = (float*)(ws + off); off += (size_t)T * B * 4 * DR * 4;    // 10.5 MB
    float* h0       = (float*)(ws + off); off += (size_t)B * DR * 4;
    float* h1       = (float*)(ws + off); off += (size_t)B * DR * 4;
    float* cbuf     = (float*)(ws + off); off += (size_t)B * DR * 4;
    ushort* hs_bf   = (ushort*)(ws + off); off += (size_t)T * B * DR * 2;
    float* bias_g   = (float*)(ws + off); off += (size_t)4 * DR * 4;
    ushort* Wb      = (ushort*)(ws + off); off += (size_t)VPAD * DR * 2;

    hipMemsetAsync(h0,   0, (size_t)B * DR * 4, stream);
    hipMemsetAsync(cbuf, 0, (size_t)B * DR * 4, stream);

    prep_bias  <<<8,    256, 0, stream>>>(b_ih, b_hh, bias_g);
    conv_wlogit<<<2048, 256, 0, stream>>>(W_logit, Wb);
    embed_gather<<<(T - 1) * B, 128, 0, stream>>>(seq, embed, xs);

    // x0 = img_feat @ W_img^T + b_img  -> xs rows [0,64)   (M=64,N=512,K=2048)
    gemm_f32_wmma<<<4, 256, 0, stream>>>(img_feat, W_img, b_img, xs,
                                         64, DIN, DF, DF, DF, DIN);
    // gates_x = xs @ W_ih^T + (b_ih+b_hh)                  (M=1280,N=2048,K=512)
    gemm_f32_wmma<<<320, 256, 0, stream>>>(xs, W_ih, bias_g, gates_x,
                                           T * B, 4 * DR, DIN, DIN, DIN, 4 * DR);

    float* hb[2] = {h0, h1};
    for (int t = 0; t < T; ++t) {
        lstm_step<<<8, 256, 0, stream>>>(hb[t & 1], cbuf,
                                         gates_x + (size_t)t * B * 4 * DR,
                                         W_hh,
                                         hb[(t + 1) & 1],
                                         hs_bf + (size_t)t * B * DR);
    }

    // strips = 80 * 501 = 40080 waves -> 5010 blocks of 8 waves
    logits_gemm<<<5010, 256, 0, stream>>>(hs_bf, Wb, b_logit, out);
    logsoftmax_rows<<<B * T, 256, 0, stream>>>(out);
}
